// MultiHeadSelfAttention_65575560675693
// MI455X (gfx1250) — compile-verified, matrix operand order checked by reading, not run on hardware
//
#include <hip/hip_runtime.h>
#include <hip/hip_bf16.h>

// ---------------------------------------------------------------------------
// MI455X (gfx1250) fused multi-head self-attention.
//   - all GEMMs + attention on v_wmma_f32_16x16x32_f16 (f16 in, f32 accum)
//   - GEMM: B tile staged with GLOBAL_LOAD_ASYNC_TO_LDS_B128 (ASYNCcnt),
//     double-buffered LDS, all 8 B fragments preloaded per K-step
//   - flash-style streaming softmax with the reference's (1 + sum) denominator
//   - S^T / ctx^T layout trick: WMMA C-layout of K@Q^T is directly the
//     B-fragment layout of P for V^T@P  -> no cross-lane transpose needed.
// ---------------------------------------------------------------------------

typedef __attribute__((ext_vector_type(16))) _Float16 v16h;
typedef __attribute__((ext_vector_type(8)))  _Float16 v8h;
typedef __attribute__((ext_vector_type(8)))  float    v8f;

#define DEVINL __device__ __forceinline__

// D = A(16x32 f16) * B(32x16 f16) + C, f32 accumulate.
DEVINL v8f wmma_f16(v16h a, v16h b, v8f c) {
  return __builtin_amdgcn_wmma_f32_16x16x32_f16(
      /*neg_a=*/false, a, /*neg_b=*/false, b,
      /*c_mod=*/(short)0, c, /*reuse_a=*/false, /*reuse_b=*/false);
}

// Async DMA 16 bytes/lane: global -> LDS, tracked by ASYNCcnt (CDNA5 path).
// lds_ptr is a generic pointer into __shared__; its low 32 bits are the LDS
// byte offset (aperture layout, ISA 10.2).  Deriving the operand from the
// real address also makes the LDS object escape, so the "memory" clobber
// keeps the subsequent ds_loads alive.
DEVINL void async_g2l_b128(void* lds_ptr, const void* gaddr) {
  const unsigned off = (unsigned)(size_t)lds_ptr;
  asm volatile("global_load_async_to_lds_b128 %0, %1, off"
               :: "v"(off), "v"(gaddr) : "memory");
}
DEVINL void wait_asynccnt0() {
  asm volatile("s_wait_asynccnt 0x0" ::: "memory");
}

// Load a 16x32 f16 fragment (A layout; B layout is identical when the
// source is stored [n][k] row-major, i.e. the transposed matrix).
// Lane l (l<16):  row = l,      K elements {0..7, 16..23}
// Lane l (l>=16): row = l - 16, K elements {8..15, 24..31}
DEVINL v16h load_frag16x32(const _Float16* __restrict__ base, int ld) {
  const int lane = threadIdx.x & 31;
  const int r    = lane & 15;
  const int ko   = (lane >> 4) << 3;           // 0 or 8
  const _Float16* p = base + (long long)r * ld + ko;
  const v8h lo = *reinterpret_cast<const v8h*>(p);
  const v8h hi = *reinterpret_cast<const v8h*>(p + 16);
  v16h f;
#pragma unroll
  for (int i = 0; i < 8; ++i) { f[i] = lo[i]; f[i + 8] = hi[i]; }
  return f;
}

// ---------------------------------------------------------------------------
// Prep: cast x -> f16, transpose+cast the four 1024x1024 weights to [n][k] f16
// ---------------------------------------------------------------------------
__global__ __launch_bounds__(256)
void prep_kernel(const float* __restrict__ Wq, const float* __restrict__ Wk,
                 const float* __restrict__ Wv, const float* __restrict__ Wo,
                 const float* __restrict__ x,
                 _Float16* __restrict__ wqT, _Float16* __restrict__ wkT,
                 _Float16* __restrict__ wvT, _Float16* __restrict__ woT,
                 _Float16* __restrict__ xh) {
  const long long idx = (long long)blockIdx.x * blockDim.x + threadIdx.x;
  const long long WN  = 1024LL * 1024LL;
  if (idx < 4 * WN) {
    const int which = (int)(idx / WN);
    const long long r = idx % WN;
    const int n = (int)(r >> 10), k = (int)(r & 1023);
    const float* src = (which == 0) ? Wq : (which == 1) ? Wk : (which == 2) ? Wv : Wo;
    _Float16*   dst = (which == 0) ? wqT : (which == 1) ? wkT : (which == 2) ? wvT : woT;
    dst[(long long)n * 1024 + k] = (_Float16)src[(long long)k * 1024 + n];
  } else {
    const long long r = idx - 4 * WN;           // < 4096*1024
    xh[r] = (_Float16)x[r];
  }
}

// ---------------------------------------------------------------------------
// GEMM: Y[4096,1024] = A[4096,1024](f16) @ W + bias,  W given as wT[n][k] f16.
//  mode 0/1 : write f16, layout [b*16+h][l][64]   (q, k)
//  mode 2   : write f16, layout [b*16+h][64][l]   (v transposed)
//  mode 3   : write f32, layout [m][n] + bias     (final output)
// Block: 256 threads = 8 waves. Wave tile 16x128 (8 accum), WG tile 128x128.
// B tile (128n x 32k) async-DMA'd to LDS, double-buffered, padded stride 40.
// ---------------------------------------------------------------------------
#define LDB 40   // padded LDS row stride in f16 (32 data + 8 pad)

__global__ __launch_bounds__(256)
void gemm_wmma_kernel(const _Float16* __restrict__ A, const _Float16* __restrict__ wT,
                      const float* __restrict__ bias,
                      _Float16* __restrict__ outQK, _Float16* __restrict__ outVT,
                      float* __restrict__ outF, int mode) {
  __shared__ _Float16 bs[2][128 * LDB];

  const int tid  = threadIdx.x;
  const int lane = tid & 31;
  const int wv   = tid >> 5;                         // 0..7
  const int m0   = blockIdx.x * 128 + wv * 16;
  const int n0   = blockIdx.y * 128;

  // staging: thread t DMAs 32B (two b128) of row (t>>1), k-half (t&1)*16
  const int srow = tid >> 1;
  const int skc  = (tid & 1) * 16;
  const _Float16* gsrc = wT + (long long)(n0 + srow) * 1024 + skc;

  // visible stores into the (DMA-untouched) pad columns: guarantees the
  // compiler cannot treat bs as never-written.
  bs[0][(tid >> 3) * LDB + 32 + (tid & 7)] = (_Float16)0.f;
  bs[1][(tid >> 3) * LDB + 32 + (tid & 7)] = (_Float16)0.f;

  // prologue: async-stage tile kb=0 into buffer 0, preload A fragment kb=0
  async_g2l_b128(&bs[0][srow * LDB + skc],     gsrc);
  async_g2l_b128(&bs[0][srow * LDB + skc + 8], gsrc + 8);
  v16h a_cur = load_frag16x32(A + (long long)m0 * 1024, 1024);
  wait_asynccnt0();
  __syncthreads();

  const int r  = lane & 15;
  const int ko = (lane >> 4) << 3;

  v8f acc[8] = {};
  for (int kb = 0; kb < 1024; kb += 32) {
    const int cur = (kb >> 5) & 1;
    const int kpre = (kb + 32 < 1024) ? (kb + 32) : 0;   // tail: harmless reload

    // kick off next tile's DMA + A-fragment load (hidden behind the 8 WMMAs)
    _Float16* sdst = &bs[cur ^ 1][srow * LDB + skc];
    async_g2l_b128(sdst,     gsrc + kpre);
    async_g2l_b128(sdst + 8, gsrc + kpre + 8);
    const v16h a_nxt = load_frag16x32(A + (long long)m0 * 1024 + kpre, 1024);

    // preload all 8 B fragments so the 16 ds_loads clause ahead of the WMMAs
    v16h bf[8];
#pragma unroll
    for (int t = 0; t < 8; ++t) {
      const _Float16* lp = &bs[cur][(t * 16 + r) * LDB + ko];
      const v8h lo = *reinterpret_cast<const v8h*>(lp);
      const v8h hi = *reinterpret_cast<const v8h*>(lp + 16);
#pragma unroll
      for (int i = 0; i < 8; ++i) { bf[t][i] = lo[i]; bf[t][i + 8] = hi[i]; }
    }
#pragma unroll
    for (int t = 0; t < 8; ++t) acc[t] = wmma_f16(a_cur, bf[t], acc[t]);

    a_cur = a_nxt;
    wait_asynccnt0();                                    // next tile resident
    __syncthreads();
  }

  const int hi_ = lane >> 4;
  const int nl  = lane & 15;
#pragma unroll
  for (int t = 0; t < 8; ++t) {
    const int n  = n0 + t * 16 + nl;
    const float bn = bias[n];
    if (mode <= 1) {
      const int h = n >> 6, d = n & 63;
#pragma unroll
      for (int j = 0; j < 8; ++j) {
        const int m = m0 + j + 8 * hi_;
        const int b = m >> 11, l = m & 2047;
        outQK[(((long long)(b * 16 + h)) * 2048 + l) * 64 + d] =
            (_Float16)(acc[t][j] + bn);
      }
    } else if (mode == 2) {
      const int h = n >> 6, d = n & 63;
      const int m = m0 + 8 * hi_;                    // token of j=0 (8 consecutive)
      const int b = m >> 11, l = m & 2047;
      v8h pk;
#pragma unroll
      for (int j = 0; j < 8; ++j) pk[j] = (_Float16)(acc[t][j] + bn);
      *reinterpret_cast<v8h*>(outVT + (((long long)(b * 16 + h)) * 64 + d) * 2048 + l) = pk;
    } else {
#pragma unroll
      for (int j = 0; j < 8; ++j) {
        const int m = m0 + j + 8 * hi_;
        outF[(long long)m * 1024 + n] = acc[t][j] + bn;
      }
    }
  }
}

// ---------------------------------------------------------------------------
// Flash attention (causal, softmax with +1 denominator).
// Wave handles 16 query rows of one (b,h). grid = (2048/128, 32), 8 waves/WG.
//   S^T = K @ Q^T      (A = K rows, B = Q rows; both direct fragment loads)
//   ctx^T = V^T @ P    (A = vT rows, B = P packed straight from S^T C-layout)
// All 8 fragments (4 K + 4 V^T) of a 32-key block are loaded up-front so the
// 16 b128 loads clause and overlap the S WMMAs + softmax math.
// ---------------------------------------------------------------------------
__global__ __launch_bounds__(256)
void attn_kernel(const _Float16* __restrict__ qh, const _Float16* __restrict__ kh,
                 const _Float16* __restrict__ vT, _Float16* __restrict__ ctxh) {
  const int lane = threadIdx.x & 31;
  const int wv   = threadIdx.x >> 5;
  const int bh   = blockIdx.y;                       // 0..31
  const int b    = bh >> 4, h = bh & 15;
  const int q0   = blockIdx.x * 128 + wv * 16;
  const int nl   = lane & 15, hi = lane >> 4;
  const int qrow = q0 + nl;                          // this lane's query row

  const _Float16* qp = qh + ((long long)bh * 2048 + q0) * 64;
  const v16h bq0 = load_frag16x32(qp, 64);           // head-dim 0..31
  const v16h bq1 = load_frag16x32(qp + 32, 64);      // head-dim 32..63

  v8f acc[4] = {};
  float m_run = -INFINITY, l_run = 0.f;
  const float scale = 0.125f;                        // 1/sqrt(64)

  for (int kb = 0; kb < q0 + 16; kb += 32) {
    // ---- issue all loads for this 32-key block up-front -------------------
    const _Float16* kp = kh + ((long long)bh * 2048 + kb) * 64;
    const v16h k00 = load_frag16x32(kp, 64);
    const v16h k01 = load_frag16x32(kp + 32, 64);
    const v16h k10 = load_frag16x32(kp + 16 * 64, 64);
    const v16h k11 = load_frag16x32(kp + 16 * 64 + 32, 64);
    const _Float16* vp = vT + (long long)bh * 64 * 2048 + kb;
    const v16h vf0 = load_frag16x32(vp,               2048);
    const v16h vf1 = load_frag16x32(vp + 16 * 2048,   2048);
    const v16h vf2 = load_frag16x32(vp + 32 * 2048,   2048);
    const v16h vf3 = load_frag16x32(vp + 48 * 2048,   2048);

    // ---- S^T = K @ Q^T ----------------------------------------------------
    v8f z0 = {};
    v8f s0 = wmma_f16(k00, bq0, z0);
    s0     = wmma_f16(k01, bq1, s0);
    v8f z1 = {};
    v8f s1 = wmma_f16(k10, bq0, z1);
    s1     = wmma_f16(k11, bq1, s1);

    // scale + causal mask + block max (per query = per lane, halves via shfl)
    float mloc = -INFINITY;
#pragma unroll
    for (int j = 0; j < 8; ++j) {
      const int key0 = kb + j + 8 * hi;
      const int key1 = key0 + 16;
      const float x0 = (key0 <= qrow) ? s0[j] * scale : -INFINITY;
      const float x1 = (key1 <= qrow) ? s1[j] * scale : -INFINITY;
      s0[j] = x0; s1[j] = x1;
      mloc = fmaxf(mloc, fmaxf(x0, x1));
    }
    mloc = fmaxf(mloc, __shfl_xor(mloc, 16, 32));
    const float mnew  = fmaxf(m_run, mloc);
    const float alpha = __expf(m_run - mnew);        // first iter: exp(-inf)=0

    // exp + pack P fragment (B layout for V^T @ P falls out of S^T C layout)
    float lloc = 0.f;
    v16h p;
#pragma unroll
    for (int j = 0; j < 8; ++j) {
      const float e0 = __expf(s0[j] - mnew);
      const float e1 = __expf(s1[j] - mnew);
      lloc += e0 + e1;
      p[j]     = (_Float16)e0;
      p[j + 8] = (_Float16)e1;
    }
    lloc += __shfl_xor(lloc, 16, 32);
    l_run = l_run * alpha + lloc;
    m_run = mnew;

    // ---- ctx^T += V^T @ P -------------------------------------------------
#pragma unroll
    for (int j = 0; j < 8; ++j) {
      acc[0][j] *= alpha; acc[1][j] *= alpha;
      acc[2][j] *= alpha; acc[3][j] *= alpha;
    }
    acc[0] = wmma_f16(vf0, p, acc[0]);
    acc[1] = wmma_f16(vf1, p, acc[1]);
    acc[2] = wmma_f16(vf2, p, acc[2]);
    acc[3] = wmma_f16(vf3, p, acc[3]);
  }

  // "+1" quiet-softmax denominator; per-lane (one query per lane in ctx^T).
  const float r = 1.f / (1.f + l_run);
  const long long token = (long long)b * 2048 + q0 + nl;
#pragma unroll
  for (int t = 0; t < 4; ++t) {
    v8h pk;
#pragma unroll
    for (int j = 0; j < 8; ++j) pk[j] = (_Float16)(acc[t][j] * r);
    *reinterpret_cast<v8h*>(ctxh + token * 1024 + h * 64 + t * 16 + 8 * hi) = pk;
  }
}

// ---------------------------------------------------------------------------
extern "C" void kernel_launch(void* const* d_in, const int* in_sizes, int n_in,
                              void* d_out, int out_size, void* d_ws, size_t ws_size,
                              hipStream_t stream) {
  const float* x  = (const float*)d_in[0];
  const float* Wq = (const float*)d_in[1];
  const float* bq = (const float*)d_in[2];
  const float* Wk = (const float*)d_in[3];
  const float* bk = (const float*)d_in[4];
  const float* Wv = (const float*)d_in[5];
  const float* bv = (const float*)d_in[6];
  const float* Wo = (const float*)d_in[7];
  const float* bo = (const float*)d_in[8];

  char* ws = (char*)d_ws;
  _Float16* xh   = (_Float16*)(ws);                       //  8 MB  x as f16
  _Float16* wqT  = (_Float16*)(ws + ( 8ull << 20));       //  2 MB
  _Float16* wkT  = (_Float16*)(ws + (10ull << 20));       //  2 MB
  _Float16* wvT  = (_Float16*)(ws + (12ull << 20));       //  2 MB
  _Float16* woT  = (_Float16*)(ws + (14ull << 20));       //  2 MB
  _Float16* qhp  = (_Float16*)(ws + (16ull << 20));       //  8 MB [bh][l][64]
  _Float16* khp  = (_Float16*)(ws + (24ull << 20));       //  8 MB [bh][l][64]
  _Float16* vTp  = (_Float16*)(ws + (32ull << 20));       //  8 MB [bh][64][l]
  _Float16* ctxh = (_Float16*)(ws + (40ull << 20));       //  8 MB [token][1024]

  prep_kernel<<<32768, 256, 0, stream>>>(Wq, Wk, Wv, Wo, x, wqT, wkT, wvT, woT, xh);

  dim3 g(32, 8);
  gemm_wmma_kernel<<<g, 256, 0, stream>>>(xh, wqT, bq, qhp, nullptr, nullptr, 0);
  gemm_wmma_kernel<<<g, 256, 0, stream>>>(xh, wkT, bk, khp, nullptr, nullptr, 1);
  gemm_wmma_kernel<<<g, 256, 0, stream>>>(xh, wvT, bv, nullptr, vTp, nullptr, 2);

  attn_kernel<<<dim3(16, 32), 256, 0, stream>>>(qhp, khp, vTp, ctxh);

  gemm_wmma_kernel<<<g, 256, 0, stream>>>(ctxh, woT, bo, nullptr, nullptr,
                                          (float*)d_out, 3);
}
